// MoELayer_6605659701904
// MI455X (gfx1250) — matrix operand order repaired and checked by reading, not compiled, not run on hardware
//
#include <hip/hip_runtime.h>
#include <hip/hip_bf16.h>

// Problem constants (match reference)
#define L_TOK   4096   // B*N
#define DM      512    // d_model
#define DFF     2048
#define NEXP    8
#define TOPK    2

typedef __attribute__((ext_vector_type(16))) _Float16 v16h;
typedef __attribute__((ext_vector_type(8)))  float    v8f;

union Frag16 { v16h h; float4 q[2]; };

// ---------------- elementwise f32 -> f16 ----------------
__global__ void f32_to_f16_kernel(const float* __restrict__ in,
                                  _Float16* __restrict__ out, int n) {
  int i = blockIdx.x * blockDim.x + threadIdx.x;
  int stride = gridDim.x * blockDim.x;
  for (; i < n; i += stride) out[i] = (_Float16)in[i];
}

// ---------------- reset per-expert counters ----------------
__global__ void reset_counts_kernel(int* counts) {
  if (threadIdx.x < NEXP) counts[threadIdx.x] = 0;
}

// ---------------- router: logits -> softmax -> top2 -> buckets ----------------
// block = 256 threads = 32 tokens x 8 experts
__global__ void router_kernel(const float* __restrict__ x,
                              const float* __restrict__ rw,
                              int* __restrict__ counts,
                              int* __restrict__ ids,
                              float* __restrict__ gates) {
  __shared__ float lg[32][NEXP];
  const int tid = threadIdx.x;
  const int tl  = tid >> 3;      // local token 0..31
  const int e   = tid & 7;       // expert 0..7
  const int token = blockIdx.x * 32 + tl;

  const float* xr = x  + (size_t)token * DM;
  const float* wr = rw + (size_t)e * DM;
  float dot = 0.f;
  #pragma unroll 8
  for (int c = 0; c < DM; ++c) dot += xr[c] * wr[c];
  lg[tl][e] = dot;
  __syncthreads();

  if (e == 0) {
    float p[NEXP];
    float mx = -1e30f;
    #pragma unroll
    for (int i = 0; i < NEXP; ++i) { float v = lg[tl][i]; mx = v > mx ? v : mx; }
    float s = 0.f;
    #pragma unroll
    for (int i = 0; i < NEXP; ++i) { p[i] = expf(lg[tl][i] - mx); s += p[i]; }
    int i0 = 0;
    #pragma unroll
    for (int i = 1; i < NEXP; ++i) if (p[i] > p[i0]) i0 = i;
    int i1 = (i0 == 0) ? 1 : 0;
    #pragma unroll
    for (int i = 0; i < NEXP; ++i) if (i != i0 && p[i] > p[i1]) i1 = i;
    const float pr0 = p[i0] / s, pr1 = p[i1] / s;
    const float inv = 1.f / (pr0 + pr1 + 1e-9f);
    int pos0 = atomicAdd(&counts[i0], 1);
    ids[i0 * L_TOK + pos0]   = token * TOPK + 0;
    gates[i0 * L_TOK + pos0] = pr0 * inv;
    int pos1 = atomicAdd(&counts[i1], 1);
    ids[i1 * L_TOK + pos1]   = token * TOPK + 1;
    gates[i1 * L_TOK + pos1] = pr1 * inv;
  }
}

// ---------------- fused expert MLP, WMMA f16 16x16x32 ----------------
// grid = (NEXP, L_TOK/16), block = 256 (8 waves). One block = one expert,
// one 16-token tile. GEMM1 and GEMM2 fused through an LDS h-tile; dff is
// processed in 2 chunks of 1024 to keep LDS at ~48KB.
__global__ __launch_bounds__(256) void moe_expert_kernel(
    const _Float16* __restrict__ xh,  const _Float16* __restrict__ w1h,
    const _Float16* __restrict__ w2h, const float* __restrict__ b1,
    const float* __restrict__ b2,     const int* __restrict__ counts,
    const int* __restrict__ ids,      const float* __restrict__ gates,
    float* __restrict__ contrib) {
  const int e    = blockIdx.x;
  const int tile = blockIdx.y;
  const int cnt  = counts[e];
  if (tile * 16 >= cnt) return;

  __shared__ _Float16 xs[16 * DM];    // 16KB  gathered x tile (f16)
  __shared__ _Float16 hs[16 * 1024];  // 32KB  h chunk tile (f16)
  __shared__ int   s_id[16];
  __shared__ float s_g[16];

  const int tid  = threadIdx.x;
  const int wave = tid >> 5;
  const int lane = tid & 31;
  const int hi   = lane >> 4;   // lane group (0: lanes 0-15, 1: lanes 16-31)
  const int lp   = lane & 15;

  if (tid < 16) {
    int row = tile * 16 + tid;
    if (row < cnt) { s_id[tid] = ids[e * L_TOK + row]; s_g[tid] = gates[e * L_TOK + row]; }
    else           { s_id[tid] = -1;                   s_g[tid] = 0.f; }
  }
  __syncthreads();

  // Gather 16 token rows of x(f16): 16 threads per row, 32 halves (64B) each.
  {
    const int r = tid >> 4, cpart = (tid & 15) * 32;
    const int pid = s_id[r];
    float4* dst = (float4*)(xs + r * DM + cpart);
    if (pid >= 0) {
      const float4* src = (const float4*)(xh + (size_t)(pid >> 1) * DM + cpart);
      #pragma unroll
      for (int i = 0; i < 4; ++i) dst[i] = src[i];
    } else {
      const float4 z = make_float4(0.f, 0.f, 0.f, 0.f);
      #pragma unroll
      for (int i = 0; i < 4; ++i) dst[i] = z;
    }
  }
  __syncthreads();

  // Persistent GEMM2 accumulators: 4 n-tiles of 16 cols per wave (512/8 waves).
  v8f acc2[4];
  {
    const v8f z = {};
    #pragma unroll
    for (int t = 0; t < 4; ++t) acc2[t] = z;
  }

  for (int chunk = 0; chunk < 2; ++chunk) {
    // ---- GEMM1: h[16, 1024-chunk] = relu(x_tile @ w1[e].T + b1) ----
    // wave covers 128 f-columns of this chunk = 8 n-tiles.
    #pragma unroll 1
    for (int nt = 0; nt < 8; ++nt) {
      const int n0l  = wave * 128 + nt * 16;          // local col in hs
      const int fcol = chunk * 1024 + n0l + lp;       // global f index
      v8f acc = {};
      // B operand: w1[f, c] row-major -> per-lane K-contiguous 32B loads.
      const _Float16* wp = w1h + ((size_t)e * DFF + fcol) * DM + hi * 16;
      #pragma unroll 4
      for (int k0 = 0; k0 < DM; k0 += 32) {
        Frag16 a, b;
        a.q[0] = *(const float4*)(xs + lp * DM + k0 + hi * 16);       // ds_load_b128
        a.q[1] = *(const float4*)(xs + lp * DM + k0 + hi * 16 + 8);
        b.q[0] = *(const float4*)(wp + k0);                            // global b128
        b.q[1] = *(const float4*)(wp + k0 + 8);
        __builtin_prefetch((const void*)(wp + k0 + 128), 0, 1);
        acc = __builtin_amdgcn_wmma_f32_16x16x32_f16(
                  false, a.h, false, b.h, (short)0, acc, false, false);
      }
      const float bias = b1[e * DFF + fcol];
      #pragma unroll
      for (int r = 0; r < 8; ++r) {
        const int m = hi * 8 + r;                     // C/D layout: m=(lane>>4)*8+r
        float v = acc[r] + bias;
        hs[m * 1024 + n0l + lp] = (_Float16)(v > 0.f ? v : 0.f);
      }
    }
    __syncthreads();

    // ---- GEMM2 partial: out += h_chunk @ w2[e][:, chunk].T ----
    #pragma unroll 1
    for (int nt = 0; nt < 4; ++nt) {
      const int n0 = wave * 64 + nt * 16;             // output channel tile
      v8f acc = acc2[nt];
      // B operand: w2[c, f] row-major -> per-lane K-contiguous 32B loads.
      const _Float16* wp = w2h + ((size_t)e * DM + n0 + lp) * DFF
                         + chunk * 1024 + hi * 16;
      #pragma unroll 4
      for (int k0 = 0; k0 < 1024; k0 += 32) {
        Frag16 a, b;
        a.q[0] = *(const float4*)(hs + lp * 1024 + k0 + hi * 16);
        a.q[1] = *(const float4*)(hs + lp * 1024 + k0 + hi * 16 + 8);
        b.q[0] = *(const float4*)(wp + k0);
        b.q[1] = *(const float4*)(wp + k0 + 8);
        __builtin_prefetch((const void*)(wp + k0 + 128), 0, 1);
        acc = __builtin_amdgcn_wmma_f32_16x16x32_f16(
                  false, a.h, false, b.h, (short)0, acc, false, false);
      }
      acc2[nt] = acc;
    }
    __syncthreads();  // hs reused by next chunk
  }

  // ---- epilogue: +b2, *gate, write per-(token,slot) contribution ----
  #pragma unroll
  for (int nt = 0; nt < 4; ++nt) {
    const int n0 = wave * 64 + nt * 16;
    const float bias = b2[e * DM + n0 + lp];
    #pragma unroll
    for (int r = 0; r < 8; ++r) {
      const int m = hi * 8 + r;
      const int pid = s_id[m];
      if (pid >= 0)
        contrib[(size_t)pid * DM + n0 + lp] = (acc2[nt][r] + bias) * s_g[m];
    }
  }
}

// ---------------- combine the two expert contributions per token ----------------
__global__ void combine_kernel(const float* __restrict__ contrib,
                               float* __restrict__ out, int n) {
  int i = blockIdx.x * blockDim.x + threadIdx.x;
  if (i >= n) return;
  const int t = i >> 9;           // token (DM = 512)
  const int c = i & (DM - 1);
  out[i] = contrib[(size_t)(t * TOPK) * DM + c]
         + contrib[(size_t)(t * TOPK + 1) * DM + c];
}

extern "C" void kernel_launch(void* const* d_in, const int* in_sizes, int n_in,
                              void* d_out, int out_size, void* d_ws, size_t ws_size,
                              hipStream_t stream) {
  (void)in_sizes; (void)n_in; (void)out_size; (void)ws_size;
  const float* x  = (const float*)d_in[0];   // [L, 512]
  const float* rw = (const float*)d_in[1];   // [8, 512]
  const float* w1 = (const float*)d_in[2];   // [8, 2048, 512]
  const float* b1 = (const float*)d_in[3];   // [8, 2048]
  const float* w2 = (const float*)d_in[4];   // [8, 512, 2048]
  const float* b2 = (const float*)d_in[5];   // [8, 512]
  float* out = (float*)d_out;                // [L, 512]

  char* ws = (char*)d_ws;
  size_t o = 0;
  int*      counts  = (int*)(ws + o);      o += 256;
  int*      ids     = (int*)(ws + o);      o += (size_t)NEXP * L_TOK * 4;
  float*    gates   = (float*)(ws + o);    o += (size_t)NEXP * L_TOK * 4;
  _Float16* xh      = (_Float16*)(ws + o); o += (size_t)L_TOK * DM * 2;
  _Float16* w1h     = (_Float16*)(ws + o); o += (size_t)NEXP * DFF * DM * 2;
  _Float16* w2h     = (_Float16*)(ws + o); o += (size_t)NEXP * DM * DFF * 2;
  float*    contrib = (float*)(ws + o);    o += (size_t)L_TOK * TOPK * DM * 4;

  reset_counts_kernel<<<1, 32, 0, stream>>>(counts);
  f32_to_f16_kernel<<<1024, 256, 0, stream>>>(x,  xh,  L_TOK * DM);
  f32_to_f16_kernel<<<2048, 256, 0, stream>>>(w1, w1h, NEXP * DFF * DM);
  f32_to_f16_kernel<<<2048, 256, 0, stream>>>(w2, w2h, NEXP * DM * DFF);
  router_kernel<<<L_TOK / 32, 256, 0, stream>>>(x, rw, counts, ids, gates);

  dim3 grid(NEXP, L_TOK / 16);
  moe_expert_kernel<<<grid, 256, 0, stream>>>(xh, w1h, w2h, b1, b2,
                                              counts, ids, gates, contrib);

  combine_kernel<<<(L_TOK * DM + 255) / 256, 256, 0, stream>>>(contrib, out, L_TOK * DM);
}
